// GRUD_37812892074641
// MI455X (gfx1250) — matrix operand order repaired and checked by reading, not compiled, not run on hardware
//
#include <hip/hip_runtime.h>
#include <hip/hip_bf16.h>
#include <math.h>

#define B_TOT    4096
#define T_LEN    512
#define H_DIM    512
#define GATE_DIM 514
#define BT       16      // batch rows per workgroup
#define SBF      520     // padded bf16 row stride (elements) to dodge bank conflicts

typedef __attribute__((ext_vector_type(16))) __bf16 v16bf;
typedef __attribute__((ext_vector_type(8)))  float  v8f;

union FragU { v16bf f; uint4 q[2]; };

// A fragment: 16x32 bf16 tile. Lane l: row = l&15; half = l>>4.
// VGPR0-3 <- K = kb + half*16 + {0..7}; VGPR4-7 <- K = kb + half*16 + 8 + {0..7}.
__device__ inline v16bf ldA(const __bf16* rowbase, int kb, int halfk) {
  FragU u;
  u.q[0] = *(const uint4*)(rowbase + kb + halfk * 16);
  u.q[1] = *(const uint4*)(rowbase + kb + halfk * 16 + 8);
  return u.f;
}

// B fragment: 32x16 bf16 tile from W stored [N][K] row-major (K stride 512).
// Lane l: col = l&15 (caller bakes N into rowbase); lanes 0-15 hold K=kb..kb+15,
// lanes 16-31 hold K=kb+16..kb+31 -> 16 contiguous bf16 per lane.
__device__ inline v16bf ldB(const __bf16* rowbase, int kb, int halfk) {
  FragU u;
  const __bf16* p = rowbase + kb + halfk * 16;
  u.q[0] = *(const uint4*)(p);
  u.q[1] = *(const uint4*)(p + 8);
  return u.f;
}

__device__ inline v8f wmma_bf16(v16bf a, v16bf b, v8f c) {
  // (neg_a, A, neg_b, B, c_mod, C, reuse_a, reuse_b)
  return __builtin_amdgcn_wmma_f32_16x16x32_bf16(false, a, false, b, (short)0, c,
                                                 false, false);
}

__device__ inline float sigmoid_f(float v) { return 1.f / (1.f + __expf(-v)); }

// ---------------------------------------------------------------------------
// Prep kernel 1: transpose-extract gate weights to bf16 [3][N=512][K=512].
// Source Wz/Wr/Wh are fp32 [H][GATE]; col 0 = x coeff, 1..512 = h block,
// col 513 = mask coeff. We keep only the h block here.
// ---------------------------------------------------------------------------
__global__ void grud_wprep(const float* __restrict__ Wz, const float* __restrict__ Wr,
                           const float* __restrict__ Wh, __bf16* __restrict__ out) {
  int row  = blockIdx.x & (H_DIM - 1);
  int gate = blockIdx.x >> 9;
  const float* src = (gate == 0 ? Wz : (gate == 1 ? Wr : Wh)) +
                     (size_t)row * GATE_DIM + 1;
  __bf16* dst = out + ((size_t)gate * H_DIM + row) * H_DIM;
  for (int c = threadIdx.x; c < H_DIM; c += 256) dst[c] = (__bf16)src[c];
}

// ---------------------------------------------------------------------------
// Prep kernel 2: x_mean[b] = sum(x*m) / sum(m). One wave32 per batch row.
// ---------------------------------------------------------------------------
__global__ void grud_xmean(const float* __restrict__ x, const float* __restrict__ mask,
                           float* __restrict__ xmean) {
  int row  = blockIdx.x * 8 + (threadIdx.x >> 5);
  int lane = threadIdx.x & 31;
  const float* xr = x + (size_t)row * T_LEN;
  const float* mr = mask + (size_t)row * T_LEN;
  float sx = 0.f, sm = 0.f;
  for (int t = lane; t < T_LEN; t += 32) {
    float m = mr[t];
    sx += xr[t] * m;
    sm += m;
  }
  for (int o = 16; o; o >>= 1) {
    sx += __shfl_xor(sx, o, 32);
    sm += __shfl_xor(sm, o, 32);
  }
  if (lane == 0) xmean[row] = sx / sm;
}

// ---------------------------------------------------------------------------
// Main recurrent kernel: 256 threads (8 wave32s) per block; block owns 16 batch
// rows; hidden state lives in LDS (bf16) for all 512 timesteps.
// Wave w computes output columns [w*64, w*64+64) as 4 WMMA n-tiles.
// ---------------------------------------------------------------------------
__global__ __launch_bounds__(256)
void grud_main(const float* __restrict__ x, const float* __restrict__ x_last,
               const float* __restrict__ interval, const float* __restrict__ mask,
               const float* __restrict__ Wgx, const float* __restrict__ bgx,
               const float* __restrict__ Wgh, const float* __restrict__ bgh,
               const float* __restrict__ Wz_full, const float* __restrict__ bz,
               const float* __restrict__ Wr_full, const float* __restrict__ br,
               const float* __restrict__ Wh_full, const float* __restrict__ bh,
               const float* __restrict__ Wo, const float* __restrict__ bo,
               const __bf16* __restrict__ Wbf,      // [3][512][512] bf16 [N][K]
               const float* __restrict__ xmean,
               float* __restrict__ out) {
  __shared__ __align__(16) __bf16 h_b[BT][SBF];    // hidden state (bf16 master)
  __shared__ __align__(16) __bf16 rh_b[BT][SBF];   // r * h staging for h_tilde
  __shared__ float gwh[H_DIM], gbh[H_DIM];         // decay layer weights
  __shared__ float xi_s[BT], it_s[BT], mt_s[BT];

  const int tid   = threadIdx.x;
  const int wave  = tid >> 5;
  const int lane  = tid & 31;
  const int r16   = lane & 15;
  const int halfk = lane >> 4;
  const int bg    = blockIdx.x * BT;

  // --- one-time init ---
  for (int idx = tid; idx < BT * H_DIM; idx += 256)
    h_b[idx >> 9][idx & (H_DIM - 1)] = (__bf16)0.f;
  for (int c = tid; c < H_DIM; c += 256) { gwh[c] = Wgh[c]; gbh[c] = bgh[c]; }

  const float wgx0 = Wgx[0], bgx0 = bgx[0];

  // per-lane column coefficients (column fixed per lane for whole run)
  float wzx_r[4], wzm_r[4], bz_r[4];
  float wrx_r[4], wrm_r[4], br_r[4];
  float whx_r[4], whm_r[4], bh_r[4];
  const __bf16* bzrow[4];
  const __bf16* brrow[4];
  const __bf16* bhrow[4];
  const __bf16* WzT = Wbf;
  const __bf16* WrT = Wbf + (size_t)H_DIM * H_DIM;
  const __bf16* WhT = Wbf + (size_t)2 * H_DIM * H_DIM;
#pragma unroll
  for (int j = 0; j < 4; ++j) {
    int col = wave * 64 + j * 16 + r16;
    wzx_r[j] = Wz_full[(size_t)col * GATE_DIM + 0];
    wzm_r[j] = Wz_full[(size_t)col * GATE_DIM + 513];
    bz_r[j]  = bz[col];
    wrx_r[j] = Wr_full[(size_t)col * GATE_DIM + 0];
    wrm_r[j] = Wr_full[(size_t)col * GATE_DIM + 513];
    br_r[j]  = br[col];
    whx_r[j] = Wh_full[(size_t)col * GATE_DIM + 0];
    whm_r[j] = Wh_full[(size_t)col * GATE_DIM + 513];
    bh_r[j]  = bh[col];
    bzrow[j] = WzT + (size_t)col * H_DIM;
    brrow[j] = WrT + (size_t)col * H_DIM;
    bhrow[j] = WhT + (size_t)col * H_DIM;
  }
  __syncthreads();

  for (int t = 0; t < T_LEN; ++t) {
    // ---- phase 0: per-row input scalars ----
    if (tid < BT) {
      int b = bg + tid;
      float xt = x[(size_t)b * T_LEN + t];
      float xl = x_last[(size_t)b * T_LEN + t];
      float it = interval[(size_t)b * T_LEN + t];
      float mt = mask[(size_t)b * T_LEN + t];
      float gx = __expf(-fmaxf(0.f, it * wgx0 + bgx0));
      float xm = xmean[b];
      xi_s[tid] = mt * xt + (1.f - mt) * (gx * xl + (1.f - gx) * xm);
      it_s[tid] = it;
      mt_s[tid] = mt;
    }
    __syncthreads();

    // ---- phase 1: gamma_h decay, h := gamma_h * h (16 thr/row, 32 cols ea) ----
    {
      int row = tid >> 4;
      int c0  = (tid & 15) * 32;
      float itr = it_s[row];
#pragma unroll 8
      for (int c = c0; c < c0 + 32; ++c) {
        float g  = __expf(-fmaxf(0.f, itr * gwh[c] + gbh[c]));
        float hv = (float)h_b[row][c] * g;
        h_b[row][c] = (__bf16)hv;
      }
    }
    __syncthreads();

    // ---- phase 2: z and r gate matmuls (shared A fragment) ----
    const __bf16* arow = &h_b[r16][0];
    v8f accz[4], accr[4];
#pragma unroll
    for (int j = 0; j < 4; ++j) {
      accz[j] = (v8f){0.f, 0.f, 0.f, 0.f, 0.f, 0.f, 0.f, 0.f};
      accr[j] = accz[j];
    }
#pragma unroll 4
    for (int k = 0; k < H_DIM; k += 32) {
      v16bf a = ldA(arow, k, halfk);
#pragma unroll
      for (int j = 0; j < 4; ++j) {
        v16bf bzf = ldB(bzrow[j], k, halfk);
        accz[j] = wmma_bf16(a, bzf, accz[j]);
        v16bf brf = ldB(brrow[j], k, halfk);
        accr[j] = wmma_bf16(a, brf, accr[j]);
        // warm the Wh tile we will need in phase 3 (L2 -> WGP$)
        __builtin_prefetch(bhrow[j] + k + halfk * 16, 0, 1);
      }
    }

    // epilogue: z -> registers, r -> rh (bf16 LDS)
    float zreg[4][8];
#pragma unroll
    for (int j = 0; j < 4; ++j) {
      int col = wave * 64 + j * 16 + r16;
#pragma unroll
      for (int i = 0; i < 8; ++i) {
        int row = i + 8 * halfk;                 // C/D layout: lane half picks M+8
        float prez = accz[j][i] + bz_r[j] + xi_s[row] * wzx_r[j] + mt_s[row] * wzm_r[j];
        zreg[j][i] = sigmoid_f(prez);
        float prer = accr[j][i] + br_r[j] + xi_s[row] * wrx_r[j] + mt_s[row] * wrm_r[j];
        float rr   = sigmoid_f(prer);
        rh_b[row][col] = (__bf16)(rr * (float)h_b[row][col]);
      }
    }
    __syncthreads();

    // ---- phase 3: h_tilde matmul over r*h, then h update ----
    const __bf16* a2row = &rh_b[r16][0];
    v8f acch[4];
#pragma unroll
    for (int j = 0; j < 4; ++j)
      acch[j] = (v8f){0.f, 0.f, 0.f, 0.f, 0.f, 0.f, 0.f, 0.f};
#pragma unroll 4
    for (int k = 0; k < H_DIM; k += 32) {
      v16bf a = ldA(a2row, k, halfk);
#pragma unroll
      for (int j = 0; j < 4; ++j) {
        v16bf bf_ = ldB(bhrow[j], k, halfk);
        acch[j] = wmma_bf16(a, bf_, acch[j]);
      }
    }
#pragma unroll
    for (int j = 0; j < 4; ++j) {
      int col = wave * 64 + j * 16 + r16;
#pragma unroll
      for (int i = 0; i < 8; ++i) {
        int row = i + 8 * halfk;
        float pre = acch[j][i] + bh_r[j] + xi_s[row] * whx_r[j] + mt_s[row] * whm_r[j];
        float ht  = tanhf(pre);
        float zv  = zreg[j][i];
        float hn  = (1.f - zv) * (float)h_b[row][col] + zv * ht;
        h_b[row][col] = (__bf16)hn;
      }
    }
    __syncthreads();
  }

  // ---- output head: out[b] = sigmoid(h . Wo + bo) ----
  {
    int row = tid >> 4;
    int seg = tid & 15;
    float p = 0.f;
    for (int c = seg * 32; c < seg * 32 + 32; ++c)
      p += (float)h_b[row][c] * Wo[c];
    p += __shfl_xor(p, 1, 32);
    p += __shfl_xor(p, 2, 32);
    p += __shfl_xor(p, 4, 32);
    p += __shfl_xor(p, 8, 32);
    if (seg == 0) out[bg + row] = sigmoid_f(p + bo[0]);
  }
}

// ---------------------------------------------------------------------------
extern "C" void kernel_launch(void* const* d_in, const int* in_sizes, int n_in,
                              void* d_out, int out_size, void* d_ws, size_t ws_size,
                              hipStream_t stream) {
  const float* x        = (const float*)d_in[0];
  const float* x_last   = (const float*)d_in[1];
  const float* interval = (const float*)d_in[2];
  const float* mask     = (const float*)d_in[3];
  const float* Wgx      = (const float*)d_in[4];
  const float* bgx      = (const float*)d_in[5];
  const float* Wgh      = (const float*)d_in[6];
  const float* bgh      = (const float*)d_in[7];
  const float* Wz       = (const float*)d_in[8];
  const float* bz       = (const float*)d_in[9];
  const float* Wr       = (const float*)d_in[10];
  const float* br       = (const float*)d_in[11];
  const float* Wh       = (const float*)d_in[12];
  const float* bh       = (const float*)d_in[13];
  const float* Wo       = (const float*)d_in[14];
  const float* bo       = (const float*)d_in[15];
  float* out            = (float*)d_out;

  __bf16* Wbf  = (__bf16*)d_ws;                                  // 3*512*512*2 B
  float* xmean = (float*)((char*)d_ws + (size_t)3 * H_DIM * H_DIM * 2);

  grud_wprep<<<3 * H_DIM, 256, 0, stream>>>(Wz, Wr, Wh, Wbf);
  grud_xmean<<<B_TOT / 8, 256, 0, stream>>>(x, mask, xmean);
  grud_main<<<B_TOT / BT, 256, 0, stream>>>(
      x, x_last, interval, mask, Wgx, bgx, Wgh, bgh, Wz, bz, Wr, br, Wh, bh,
      Wo, bo, Wbf, xmean, out);
}
